// Attention_9577777070530
// MI455X (gfx1250) — matrix operand order repaired and checked by reading, not compiled, not run on hardware
//
#include <hip/hip_runtime.h>
#include <stdint.h>

// ---------------------------------------------------------------------------
// CDNA5 (gfx1250) bf16 WMMA attention block.
//   x -> LayerNorm -> Q = xn@Wq, KV = xn@Wkv -> causal flash attention
//     -> O @ Wo  (f32 out)
// Matrix work: V_WMMA_F32_16X16X32_BF16, f32 accumulate.
// Tile staging: GLOBAL_LOAD_ASYNC_TO_LDS_B128 (ASYNCcnt) when available.
// ---------------------------------------------------------------------------

typedef __attribute__((ext_vector_type(16))) __bf16 v16bf;
typedef __attribute__((ext_vector_type(8)))  __bf16 v8bf;
typedef __attribute__((ext_vector_type(8)))  float  v8f;

#if defined(__has_builtin)
#if __has_builtin(__builtin_amdgcn_global_load_async_to_lds_b128)
#define HAS_ASYNC_LDS 1
#endif
#endif

// Pointer types matching the async builtin's signature:
// arg0: AS(1) int __vector(4)*  (global source)
// arg1: AS(3) int __vector(4)*  (LDS destination)
typedef int v4i_vs __attribute__((vector_size(16)));
typedef __attribute__((address_space(1))) v4i_vs* as1_v4i;
typedef __attribute__((address_space(3))) v4i_vs* as3_v4i;

// 16-byte global -> LDS copy. Async DMA path on gfx1250 (no VGPR round trip,
// tracked by ASYNCcnt); falls back to a plain vector copy.
__device__ __forceinline__ void copy16_g2l(__bf16* ldst, const __bf16* gsrc) {
#ifdef HAS_ASYNC_LDS
  __builtin_amdgcn_global_load_async_to_lds_b128((as1_v4i)gsrc, (as3_v4i)ldst,
                                                 0, 0);
#else
  *(v8bf*)ldst = *(const v8bf*)gsrc;
#endif
}

__device__ __forceinline__ void wait_async() {
#ifdef HAS_ASYNC_LDS
#if __has_builtin(__builtin_amdgcn_s_wait_asynccnt)
  __builtin_amdgcn_s_wait_asynccnt(0);
#else
  asm volatile("s_wait_asynccnt 0x0" ::: "memory");
#endif
#endif
}

__device__ __forceinline__ v8f wmma_bf16(v16bf a, v16bf b, v8f c) {
  // 8 args: (neg_a, A, neg_b, B, c_mod, C, reuse_a, reuse_b)
  return __builtin_amdgcn_wmma_f32_16x16x32_bf16(
      false, a, false, b, (short)0, c, false, false);
}

// A fragment (16x32 bf16, row-major tile in LDS, leading dim ldk):
// lane holds row (lane&15); elements are K = base..base+7, base+16..base+23
// with base = (lane>>4)*8  (ISA 16-bit A 16x32 layout).
__device__ __forceinline__ v16bf load_frag_a(const __bf16* tile, int ldk,
                                             int kbase, int lane) {
  const __bf16* p = tile + (size_t)(lane & 15) * ldk + kbase + ((lane >> 4) << 3);
  v8bf lo = *(const v8bf*)p;
  v8bf hi = *(const v8bf*)(p + 16);
  v16bf r;
#pragma unroll
  for (int i = 0; i < 8; ++i) { r[i] = lo[i]; r[i + 8] = hi[i]; }
  return r;
}

// B fragment (32x16 bf16). Tile stored COLUMN-major in LDS (column c = ldk
// contiguous K values). Lane holds column (lane&15), K = base..base+7,
// base+16..base+23 with base = kbase+(lane>>4)*8 (matches A fragment order).
__device__ __forceinline__ v16bf load_frag_b(const __bf16* tile, int ldk,
                                             int kbase, int lane) {
  const __bf16* p = tile + (size_t)(lane & 15) * ldk + kbase + ((lane >> 4) << 3);
  v8bf lo = *(const v8bf*)p;
  v8bf hi = *(const v8bf*)(p + 16);
  v16bf r;
#pragma unroll
  for (int i = 0; i < 8; ++i) { r[i] = lo[i]; r[i + 8] = hi[i]; }
  return r;
}

// ---------------------------------------------------------------------------
// LayerNorm: one 256-thread block per row of 1024 floats -> bf16 out.
// ---------------------------------------------------------------------------
__global__ void __launch_bounds__(256)
ln_kernel(const float* __restrict__ x, const float* __restrict__ g,
          __bf16* __restrict__ xn) {
  constexpr int D = 1024;
  __shared__ float s1[256];
  __shared__ float s2[256];
  const int row = blockIdx.x;
  const int tid = threadIdx.x;

  const float4 xv = *(const float4*)&x[(size_t)row * D + tid * 4];
  s1[tid] = xv.x + xv.y + xv.z + xv.w;
  s2[tid] = xv.x * xv.x + xv.y * xv.y + xv.z * xv.z + xv.w * xv.w;
  __syncthreads();
#pragma unroll
  for (int s = 128; s > 0; s >>= 1) {
    if (tid < s) { s1[tid] += s1[tid + s]; s2[tid] += s2[tid + s]; }
    __syncthreads();
  }
  const float mu   = s1[0] * (1.0f / D);
  const float var  = s2[0] * (1.0f / D) - mu * mu;
  const float rstd = rsqrtf(var + 1e-5f);
  const float4 gv  = *(const float4*)&g[tid * 4];
  __bf16* o = &xn[(size_t)row * D + tid * 4];
  o[0] = (__bf16)((xv.x - mu) * rstd * gv.x);
  o[1] = (__bf16)((xv.y - mu) * rstd * gv.y);
  o[2] = (__bf16)((xv.z - mu) * rstd * gv.z);
  o[3] = (__bf16)((xv.w - mu) * rstd * gv.w);
}

// ---------------------------------------------------------------------------
// Weight transpose + f32->bf16: W[K][Nc] -> Wt[Nc][K] (bf16).
// Makes every GEMM B-tile a set of contiguous 16-byte chunks.
// ---------------------------------------------------------------------------
__global__ void __launch_bounds__(256)
wtrans_kernel(const float* __restrict__ W, __bf16* __restrict__ Wt,
              int K, int Nc) {
  __shared__ float t[64][65];
  const int n0 = blockIdx.x * 64;
  const int k0 = blockIdx.y * 64;
  const int tid = threadIdx.x;
#pragma unroll
  for (int i = tid; i < 64 * 64; i += 256) {
    const int kk = i >> 6, nn = i & 63;
    t[kk][nn] = W[(size_t)(k0 + kk) * Nc + n0 + nn];
  }
  __syncthreads();
#pragma unroll
  for (int i = tid; i < 64 * 64; i += 256) {
    const int nn = i >> 6, kk = i & 63;
    Wt[(size_t)(n0 + nn) * K + k0 + kk] = (__bf16)t[kk][nn];
  }
}

// ---------------------------------------------------------------------------
// V transpose: kvb[(b*N+n)*2D + D + h*64 + dh] -> Vt[(b*H+h)*64 + dh][N].
// Makes attention V tiles contiguous b128 chunks.
// ---------------------------------------------------------------------------
__global__ void __launch_bounds__(256)
vtrans_kernel(const __bf16* __restrict__ KVb, __bf16* __restrict__ Vt) {
  constexpr int N = 2048, D = 1024, H = 16;
  __shared__ __bf16 t[64][66];
  const int n0 = blockIdx.x * 64;
  const int h  = blockIdx.y;
  const int b  = blockIdx.z;
  const int tid = threadIdx.x;
  const __bf16* src = KVb + (size_t)(b * N) * (2 * D) + D + h * 64;
#pragma unroll
  for (int i = tid; i < 64 * 32; i += 256) {
    const int nn = i >> 5, d2 = (i & 31) << 1;
    *(uint32_t*)&t[nn][d2] =
        *(const uint32_t*)&src[(size_t)(n0 + nn) * (2 * D) + d2];
  }
  __syncthreads();
  __bf16* dst = Vt + (size_t)(b * H + h) * 64 * N;
#pragma unroll
  for (int i = tid; i < 64 * 64; i += 256) {
    const int dd = i >> 6, nn = i & 63;
    dst[(size_t)dd * N + n0 + nn] = t[nn][dd];
  }
}

// ---------------------------------------------------------------------------
// bf16 GEMM: C[M,Nc] = A[M,K] @ Bt^T where Bt is [Nc][K] row-major bf16.
// 256 threads = 8 waves; block tile 128x64; wave tile 32x32 (4 accumulators,
// 4 WMMAs per 32-wide K step). Tiles staged via async LDS DMA.
// ---------------------------------------------------------------------------
__global__ void __launch_bounds__(256)
gemm_bf16_kernel(const __bf16* __restrict__ A, const __bf16* __restrict__ Bt,
                 __bf16* __restrict__ Cb, float* __restrict__ Cf,
                 int M, int K, int Nc) {
  __shared__ __bf16 As[128][32];  // row-major [row][k]
  __shared__ __bf16 Bs[64][32];   // col-major [col][k]  (= rows of Bt)

  const int tid  = threadIdx.x;
  const int lane = tid & 31;
  const int w    = tid >> 5;
  const int wm   = w >> 1;   // 0..3 -> 32-row slab
  const int wn   = w & 1;    // 0..1 -> 32-col slab
  const int n0   = blockIdx.x * 64;
  const int row0 = blockIdx.y * 128;

  const v8f vzero = {0.f, 0.f, 0.f, 0.f, 0.f, 0.f, 0.f, 0.f};
  v8f acc00 = vzero, acc01 = vzero, acc10 = vzero, acc11 = vzero;

  for (int k0 = 0; k0 < K; k0 += 32) {
    __syncthreads();
    // A tile: 128x32 bf16 = 512 x 16B chunks; 2 per thread
#pragma unroll
    for (int c = tid; c < 512; c += 256) {
      const int r = c >> 2, ko = (c & 3) << 3;
      copy16_g2l(&As[r][ko], &A[(size_t)(row0 + r) * K + k0 + ko]);
    }
    // B tile: 64x32 bf16 = 256 x 16B chunks; 1 per thread (contiguous rows of Bt)
    {
      const int col = tid >> 2, ko = (tid & 3) << 3;
      copy16_g2l(&Bs[col][ko], &Bt[(size_t)(n0 + col) * K + k0 + ko]);
    }
    wait_async();
    __syncthreads();

    const v16bf a0 = load_frag_a(&As[wm * 32][0],      32, 0, lane);
    const v16bf a1 = load_frag_a(&As[wm * 32 + 16][0], 32, 0, lane);
    const v16bf b0 = load_frag_b(&Bs[wn * 32][0],      32, 0, lane);
    const v16bf b1 = load_frag_b(&Bs[wn * 32 + 16][0], 32, 0, lane);
    acc00 = wmma_bf16(a0, b0, acc00);
    acc01 = wmma_bf16(a0, b1, acc01);
    acc10 = wmma_bf16(a1, b0, acc10);
    acc11 = wmma_bf16(a1, b1, acc11);
  }

  // C layout: VGPR r -> row = 8*(lane>=16)+r, col = lane&15
  const int rbase = (lane >> 4) << 3;
  const int cl    = lane & 15;
  const int r0g   = row0 + wm * 32 + rbase;
  const int c0g   = n0 + wn * 32 + cl;
#pragma unroll
  for (int r = 0; r < 8; ++r) {
    if (Cf) {
      Cf[(size_t)(r0g + r) * Nc + c0g]           = acc00[r];
      Cf[(size_t)(r0g + r) * Nc + c0g + 16]      = acc01[r];
      Cf[(size_t)(r0g + 16 + r) * Nc + c0g]      = acc10[r];
      Cf[(size_t)(r0g + 16 + r) * Nc + c0g + 16] = acc11[r];
    } else {
      Cb[(size_t)(r0g + r) * Nc + c0g]           = (__bf16)acc00[r];
      Cb[(size_t)(r0g + r) * Nc + c0g + 16]      = (__bf16)acc01[r];
      Cb[(size_t)(r0g + 16 + r) * Nc + c0g]      = (__bf16)acc10[r];
      Cb[(size_t)(r0g + 16 + r) * Nc + c0g + 16] = (__bf16)acc11[r];
    }
  }
}

// ---------------------------------------------------------------------------
// Causal flash attention. One 128-thread block (4 waves) per
// (b, h, 64-row query tile). Each wave owns 16 query rows x 64 key cols.
// Q: Qb[(b*N+n)*D + h*64 + dh]
// K: KVb[(b*N+n)*2D + h*64 + dh]
// V: Vt[((b*H+h)*64 + dh)*N + n]   (pre-transposed)
// ---------------------------------------------------------------------------
__global__ void __launch_bounds__(128)
attn_kernel(const __bf16* __restrict__ Qb, const __bf16* __restrict__ KVb,
            const __bf16* __restrict__ Vt, __bf16* __restrict__ Ob) {
  constexpr int N = 2048, D = 1024, H = 16;
  __shared__ __bf16 Qs[64][64];     // [row][dh]   (A source)
  __shared__ __bf16 Ks[64][64];     // [key][dh]   (B source: col=key, k=dh)
  __shared__ __bf16 Vs[64][64];     // [dh][key]   (B source: col=dh, k=key)
  __shared__ __bf16 Ps[4][16][64];  // per-wave P  (A source)

  const int tid   = threadIdx.x;
  const int lane  = tid & 31;
  const int w     = tid >> 5;
  const int itile = blockIdx.x;
  const int h     = blockIdx.y;
  const int b     = blockIdx.z;
  const int i0    = itile * 64;

  const __bf16* Qg  = Qb  + (size_t)(b * N) * D       + (size_t)h * 64;
  const __bf16* Kg  = KVb + (size_t)(b * N) * (2 * D) + (size_t)h * 64;
  const __bf16* Vtg = Vt  + (size_t)(b * H + h) * 64 * N;

  // Q tile: 64x64 bf16 = 512 x 16B chunks; 4 per thread (async)
#pragma unroll
  for (int c = tid; c < 512; c += 128) {
    const int r = c >> 3, ko = (c & 7) << 3;
    copy16_g2l(&Qs[r][ko], &Qg[(size_t)(i0 + r) * D + ko]);
  }

  const v8f vzero = {0.f, 0.f, 0.f, 0.f, 0.f, 0.f, 0.f, 0.f};
  v8f acc[4];
  float mrow[8], lrow[8];
#pragma unroll
  for (int f = 0; f < 4; ++f) acc[f] = vzero;
#pragma unroll
  for (int r = 0; r < 8; ++r) { mrow[r] = -3.0e38f; lrow[r] = 0.0f; }

  const int   rbase = (lane >> 4) << 3;  // 0 or 8
  const int   cl    = lane & 15;
  const float scale = 0.125f;            // DH^-0.5, DH=64

  for (int jt = 0; jt <= itile; ++jt) {
    const int j0 = jt * 64;
    __syncthreads();  // previous iteration's Ks/Vs readers are done
#pragma unroll
    for (int c = tid; c < 512; c += 128) {        // K tile (row-major rows)
      const int r = c >> 3, ko = (c & 7) << 3;
      copy16_g2l(&Ks[r][ko], &Kg[(size_t)(j0 + r) * (2 * D) + ko]);
    }
#pragma unroll
    for (int c = tid; c < 512; c += 128) {        // V tile (pre-transposed rows)
      const int d = c >> 3, ko = (c & 7) << 3;
      copy16_g2l(&Vs[d][ko], &Vtg[(size_t)d * N + j0 + ko]);
    }
    wait_async();
    __syncthreads();

    // ---- S = Q @ K^T (16 x 64 per wave) ----
    v8f s[4];
#pragma unroll
    for (int f = 0; f < 4; ++f) s[f] = vzero;
#pragma unroll
    for (int kk = 0; kk < 64; kk += 32) {
      const v16bf aq = load_frag_a(&Qs[w * 16][0], 64, kk, lane);
#pragma unroll
      for (int f = 0; f < 4; ++f) {
        const v16bf bk = load_frag_b(&Ks[f * 16][0], 64, kk, lane);
        s[f] = wmma_bf16(aq, bk, s[f]);
      }
    }

    // ---- online softmax over 64 key columns ----
    const bool diag = (jt == itile);
#pragma unroll
    for (int r = 0; r < 8; ++r) {
      const int gi = i0 + w * 16 + rbase + r;
      float rm = -3.0e38f;
#pragma unroll
      for (int f = 0; f < 4; ++f) {
        float v = s[f][r] * scale;
        if (diag && (j0 + f * 16 + cl) > gi) v = -3.0e38f;
        s[f][r] = v;
        rm = fmaxf(rm, v);
      }
#pragma unroll
      for (int off = 1; off < 16; off <<= 1)
        rm = fmaxf(rm, __shfl_xor(rm, off, 32));
      const float mo    = mrow[r];
      const float mn    = fmaxf(mo, rm);
      const float alpha = __expf(mo - mn);
      float rs = 0.0f;
#pragma unroll
      for (int f = 0; f < 4; ++f) {
        const float p = __expf(s[f][r] - mn);
        s[f][r] = p;
        rs += p;
      }
#pragma unroll
      for (int off = 1; off < 16; off <<= 1)
        rs += __shfl_xor(rs, off, 32);
      lrow[r] = lrow[r] * alpha + rs;
      mrow[r] = mn;
#pragma unroll
      for (int f = 0; f < 4; ++f) {
        acc[f][r] *= alpha;
        // C-layout -> row-major LDS so P reloads as A fragments
        Ps[w][rbase + r][f * 16 + cl] = (__bf16)s[f][r];
      }
    }

    // ---- O += P @ V ----
#pragma unroll
    for (int kk = 0; kk < 64; kk += 32) {
      const v16bf ap = load_frag_a(&Ps[w][0][0], 64, kk, lane);
#pragma unroll
      for (int f = 0; f < 4; ++f) {
        const v16bf bv = load_frag_b(&Vs[f * 16][0], 64, kk, lane);
        acc[f] = wmma_bf16(ap, bv, acc[f]);
      }
    }
  }

  // ---- normalize and store O (bf16, [B*N, D] with head offset) ----
#pragma unroll
  for (int r = 0; r < 8; ++r) {
    const float  inv  = 1.0f / lrow[r];
    const size_t rowo =
        (size_t)(b * N + i0 + w * 16 + rbase + r) * D + (size_t)h * 64;
#pragma unroll
    for (int f = 0; f < 4; ++f)
      Ob[rowo + f * 16 + cl] = (__bf16)(acc[f][r] * inv);
  }
}

// ---------------------------------------------------------------------------
// Host-side orchestration.
// ---------------------------------------------------------------------------
extern "C" void kernel_launch(void* const* d_in, const int* in_sizes, int n_in,
                              void* d_out, int out_size, void* d_ws,
                              size_t ws_size, hipStream_t stream) {
  (void)in_sizes; (void)n_in; (void)out_size; (void)ws_size;
  constexpr int B = 4, N = 2048, D = 1024, H = 16;

  const float* x   = (const float*)d_in[0];
  const float* gln = (const float*)d_in[1];
  const float* Wq  = (const float*)d_in[2];
  const float* Wkv = (const float*)d_in[3];
  const float* Wo  = (const float*)d_in[4];
  float* out = (float*)d_out;

  char*  ws  = (char*)d_ws;
  size_t off = 0;
  auto alloc = [&](size_t bytes) -> void* {
    void* p = ws + off;
    off += (bytes + 255) & ~(size_t)255;
    return p;
  };
  __bf16* xn   = (__bf16*)alloc((size_t)B * N * D * 2);      // 16 MB
  __bf16* wqt  = (__bf16*)alloc((size_t)D * D * 2);          //  2 MB
  __bf16* wkvt = (__bf16*)alloc((size_t)D * 2 * D * 2);      //  4 MB
  __bf16* wot  = (__bf16*)alloc((size_t)D * D * 2);          //  2 MB
  __bf16* qb   = (__bf16*)alloc((size_t)B * N * D * 2);      // 16 MB
  __bf16* kvb  = (__bf16*)alloc((size_t)B * N * 2 * D * 2);  // 32 MB
  __bf16* vtb  = (__bf16*)alloc((size_t)B * N * D * 2);      // 16 MB
  __bf16* ob   = (__bf16*)alloc((size_t)B * N * D * 2);      // 16 MB

  // 1) LayerNorm -> bf16 activations
  ln_kernel<<<B * N, 256, 0, stream>>>(x, gln, xn);

  // 2) Weights -> bf16, transposed to [Nc][K]
  wtrans_kernel<<<dim3(D / 64, D / 64), 256, 0, stream>>>(Wq, wqt, D, D);
  wtrans_kernel<<<dim3(2 * D / 64, D / 64), 256, 0, stream>>>(Wkv, wkvt, D, 2 * D);
  wtrans_kernel<<<dim3(D / 64, D / 64), 256, 0, stream>>>(Wo, wot, D, D);

  // 3) Q = xn @ Wq ; KV = xn @ Wkv   (WMMA bf16)
  gemm_bf16_kernel<<<dim3(D / 64, (B * N) / 128), 256, 0, stream>>>(
      xn, wqt, qb, nullptr, B * N, D, D);
  gemm_bf16_kernel<<<dim3((2 * D) / 64, (B * N) / 128), 256, 0, stream>>>(
      xn, wkvt, kvb, nullptr, B * N, D, 2 * D);

  // 4) V -> [B,H,DH,N] transposed copy for contiguous attention B-tiles
  vtrans_kernel<<<dim3(N / 64, H, B), 256, 0, stream>>>(kvb, vtb);

  // 5) causal flash attention (WMMA bf16, f32 softmax)
  attn_kernel<<<dim3(N / 64, H, B), 128, 0, stream>>>(qb, kvb, vtb, ob);

  // 6) out = O @ Wo (f32 output)
  gemm_bf16_kernel<<<dim3(D / 64, (B * N) / 128), 256, 0, stream>>>(
      ob, wot, nullptr, out, B * N, D, D);
}
